// GRAPEIso_51015621542647
// MI455X (gfx1250) — matrix operand order repaired and checked by reading, not compiled, not run on hardware
//
#include <hip/hip_runtime.h>

typedef __attribute__((ext_vector_type(16))) _Float16 v16h;
typedef __attribute__((ext_vector_type(8)))  float    v8f;

#define HIN   128          // input spatial
#define HOUT  256          // output spatial
#define NPIX_OUT (HOUT*HOUT)

// ---------------------------------------------------------------------------
// Hardware tanh: gfx1250 has V_TANH_F32 (TRANS32 op, co-executes with VALU).
// Prefer the builtin (compiler handles the TRANS data hazard); fall back to
// inline asm with a trailing v_nop to satisfy the "1 independent op after
// TRANS before result use" rule the compiler can't see through asm.
// ---------------------------------------------------------------------------
__device__ __forceinline__ float hw_tanhf(float x) {
#if __has_builtin(__builtin_amdgcn_tanhf)
    return __builtin_amdgcn_tanhf(x);
#else
    float r;
    asm volatile("v_tanh_f32 %0, %1\n\tv_nop" : "=v"(r) : "v"(x));
    return r;
#endif
}

// ---------------------------------------------------------------------------
// Kernel 0: zero the output image (harness poisons d_out; atomics accumulate).
// 786432 floats = 196608 float4 -> 768 blocks x 256 threads.
// ---------------------------------------------------------------------------
__global__ __launch_bounds__(256) void GRAPE_zero_k(float4* __restrict__ out) {
    out[blockIdx.x * 256 + threadIdx.x] = make_float4(0.f, 0.f, 0.f, 0.f);
}

// ---------------------------------------------------------------------------
// Kernel 1: fuse head GEMM into the conv weights.
//   Wf[n=(parity*24+o)][k27 padded to 32] (f16)  = sum_of w_head[o,of]*w_enc[sub][k]
//   bias[n] = b_head[o] + sum_of w_head[o,of]*b_enc[sub]
// sub(parity=(py,px), of=c*4+s1*2+s2) = c*16 + (2*py+s1)*4 + (2*px+s2)
// grid: 96 blocks x 32 threads (thread = k).
// ---------------------------------------------------------------------------
__global__ __launch_bounds__(32) void GRAPE_fusew_k(const float* __restrict__ w_enc,
                                                    const float* __restrict__ b_enc,
                                                    const float* __restrict__ w_head,
                                                    const float* __restrict__ b_head,
                                                    _Float16* __restrict__ Wf,
                                                    float* __restrict__ bias) {
    int n  = blockIdx.x;           // 0..95
    int k  = threadIdx.x;          // 0..31
    int par = n / 24, o = n - 24 * par;
    int py = par >> 1, px = par & 1;

    if (k < 27) {
        float acc = 0.f;
        for (int of = 0; of < 256; ++of) {
            int c  = of >> 2;
            int s1 = (of >> 1) & 1;
            int s2 = of & 1;
            int ch = c * 16 + (2 * py + s1) * 4 + (2 * px + s2);
            acc += w_head[o * 256 + of] * w_enc[ch * 27 + k];
        }
        Wf[n * 32 + k] = (_Float16)acc;
    } else {
        Wf[n * 32 + k] = (_Float16)0.f;   // K padding 27..31 -> zeros
    }
    if (k == 31) {
        float bb = b_head[o];
        for (int of = 0; of < 256; ++of) {
            int c  = of >> 2;
            int s1 = (of >> 1) & 1;
            int s2 = of & 1;
            int ch = c * 16 + (2 * py + s1) * 4 + (2 * px + s2);
            bb += w_head[o * 256 + of] * b_enc[ch];
        }
        bias[n] = bb;
    }
}

// ---------------------------------------------------------------------------
// Kernel 2: fused implicit-GEMM conv (WMMA f16->f32) + gaussian splatting.
// One wave = 16 conv pixels (one h row segment) x 96 fused outputs
//          = 64 output pixels (2x2 parity block per conv pixel).
// Grid: 512 blocks x 256 threads (8 waves) = 4096 waves = 4*128*8 tiles.
// ---------------------------------------------------------------------------
__global__ __launch_bounds__(256) void GRAPE_conv_raster_k(const float* __restrict__ inp,
                                                           const _Float16* __restrict__ Wf,
                                                           const float* __restrict__ bias,
                                                           float* __restrict__ out) {
    __shared__ float lds[8 * 16 * 96];          // 48 KB: per-wave 16 pixels x 96 ch

    const int lane = threadIdx.x & 31;
    const int wv   = threadIdx.x >> 5;
    const int gwv  = blockIdx.x * 8 + wv;       // 0..4095
    const int b    = gwv >> 10;                 // 1024 waves per batch image
    const int rem  = gwv & 1023;
    const int h    = rem >> 3;                  // conv row 0..127
    const int w0   = (rem & 7) << 4;            // conv col tile base

    // ---- A fragment: 16x32 f16, contiguous K order (K = 16*(lane/16) + v) -------------------------
    // lanes 0-15: M=lane,   K in {0..15}
    // lanes16-31: M=lane-16, K in {16..31}
    const int M     = lane & 15;
    const int khalf = (lane >> 4) * 8;
    const int wpix  = w0 + M;
    const float* inb = inp + (size_t)b * 3 * HIN * HIN;

    v16h afrag;
#pragma unroll
    for (int v = 0; v < 16; ++v) {
        int K = 2 * khalf + v;
        float val = 0.f;
        if (K < 27) {
            int c  = K / 9;
            int r9 = K - 9 * c;
            int ky = r9 / 3;
            int kx = r9 - 3 * ky;
            int ih = h + ky - 1;
            int iw = wpix + kx - 1;
            if (ih >= 0 && ih < HIN && iw >= 0 && iw < HIN)
                val = inb[(c * HIN + ih) * HIN + iw];
        }
        afrag[v] = (_Float16)val;
    }

    // ---- 6 N-tiles of WMMA: D = A x B + bias ----------------------------
    float* myLds = lds + wv * 16 * 96;
    const int N  = lane & 15;                   // B/C/D column for this lane
    const int kb = (lane >> 4) << 4;            // B K-offset: 0 or 16
    const int mbase = (lane >> 4) * 8;          // D rows this lane holds

#pragma unroll
    for (int nt = 0; nt < 6; ++nt) {
        int ng = nt * 16 + N;                   // fused output channel 0..95
        v16h bfrag = *(const v16h*)(Wf + ng * 32 + kb);
        float bv = bias[ng];
        v8f c = {bv, bv, bv, bv, bv, bv, bv, bv};
        v8f d = __builtin_amdgcn_wmma_f32_16x16x32_f16(
            /*neg_a=*/false, afrag, /*neg_b=*/false, bfrag,
            /*c_mod=*/(short)0, c, /*reuse_a=*/false, /*reuse_b=*/false);
#pragma unroll
        for (int r = 0; r < 8; ++r)
            myLds[(mbase + r) * 96 + ng] = d[r];
    }

    __syncthreads();   // order LDS writes before per-pixel re-gather

    // ---- Gaussian raster: 64 output pixels per wave ---------------------
    float* img = out + (size_t)b * 3 * NPIX_OUT;
#pragma unroll
    for (int t = 0; t < 2; ++t) {
        int opix = t * 32 + lane;               // 0..63
        int m    = opix >> 2;                   // conv pixel 0..15
        int par  = opix & 3;                    // parity
        int py   = par >> 1, pxp = par & 1;
        const float* g = myLds + m * 96 + par * 24;   // [kg*6+d]

        // softmax over the 4 gaussian logits (d==5)
        float l0 = g[5], l1 = g[11], l2 = g[17], l3 = g[23];
        float mx = fmaxf(fmaxf(l0, l1), fmaxf(l2, l3));
        float e0 = __expf(l0 - mx), e1 = __expf(l1 - mx);
        float e2 = __expf(l2 - mx), e3 = __expf(l3 - mx);
        float inv = 1.f / (e0 + e1 + e2 + e3);
        float q0 = e0 * inv, q1 = e1 * inv, q2 = e2 * inv, q3 = e3 * inv;

        float cr = q0 * g[0] + q1 * g[6]  + q2 * g[12] + q3 * g[18];
        float cg = q0 * g[1] + q1 * g[7]  + q2 * g[13] + q3 * g[19];
        float cb = q0 * g[2] + q1 * g[8]  + q2 * g[14] + q3 * g[20];
        float ox = q0 * hw_tanhf(g[3]) + q1 * hw_tanhf(g[9])  +
                   q2 * hw_tanhf(g[15]) + q3 * hw_tanhf(g[21]);
        float oy = q0 * hw_tanhf(g[4]) + q1 * hw_tanhf(g[10]) +
                   q2 * hw_tanhf(g[16]) + q3 * hw_tanhf(g[22]);

        int H2 = 2 * h + py;
        int W2 = 2 * (w0 + m) + pxp;
        // NDC per reference: coord + 2*off/h_in - 1/H_out
        float xndc = (float)W2 * (2.0f / 256.0f) - 1.0f + 2.0f * ox / 128.0f - (1.0f / 256.0f);
        float yndc = (float)H2 * (2.0f / 256.0f) - 1.0f + 2.0f * oy / 128.0f - (1.0f / 256.0f);
        float cx = 0.5f * ((xndc + 1.0f) * 256.0f - 1.0f);
        float cy = 0.5f * ((yndc + 1.0f) * 256.0f - 1.0f);
        float fx = floorf(cx), fy = floorf(cy);

#pragma unroll
        for (int jy = 0; jy < 5; ++jy) {
            float pyf = fy + (float)(jy - 2);
            int   pyi = (int)pyf;
            float dy  = pyf - cy;
            float dy2 = dy * dy;
            bool yok  = (pyi >= 0) && (pyi < HOUT);
#pragma unroll
            for (int jx = 0; jx < 5; ++jx) {
                float pxf = fx + (float)(jx - 2);
                int   pxi = (int)pxf;
                float dx  = pxf - cx;
                // power = -0.5*(dx^2+dy^2)/s_px^2, s_px=0.5 -> -2*(..)
                float alpha = fminf(__expf(-2.0f * (dx * dx + dy2)), 0.999f);
                if (alpha > (1.0f / 255.0f) && yok && pxi >= 0 && pxi < HOUT) {
                    float* p = img + pyi * HOUT + pxi;
                    atomicAdd(p,                 alpha * cr);
                    atomicAdd(p + NPIX_OUT,      alpha * cg);
                    atomicAdd(p + 2 * NPIX_OUT,  alpha * cb);
                }
            }
        }
    }
}

// ---------------------------------------------------------------------------
// Kernel 3: final clamp to [0,1].
// ---------------------------------------------------------------------------
__global__ __launch_bounds__(256) void GRAPE_clamp_k(float4* __restrict__ out) {
    float4 v = out[blockIdx.x * 256 + threadIdx.x];
    v.x = fminf(fmaxf(v.x, 0.f), 1.f);
    v.y = fminf(fmaxf(v.y, 0.f), 1.f);
    v.z = fminf(fmaxf(v.z, 0.f), 1.f);
    v.w = fminf(fmaxf(v.w, 0.f), 1.f);
    out[blockIdx.x * 256 + threadIdx.x] = v;
}

// ---------------------------------------------------------------------------
extern "C" void kernel_launch(void* const* d_in, const int* in_sizes, int n_in,
                              void* d_out, int out_size, void* d_ws, size_t ws_size,
                              hipStream_t stream) {
    const float* inp    = (const float*)d_in[0];   // (4,3,128,128)
    const float* w_enc  = (const float*)d_in[1];   // (1024,3,3,3)
    const float* b_enc  = (const float*)d_in[2];   // (1024,)
    const float* w_head = (const float*)d_in[3];   // (24,256)
    const float* b_head = (const float*)d_in[4];   // (24,)
    (void)in_sizes; (void)n_in; (void)out_size;    // scale hardcoded to 2

    float*     out  = (float*)d_out;               // (4,3,256,256)
    _Float16*  Wf   = (_Float16*)d_ws;             // [96][32] f16   (6144 B)
    float*     bias = (float*)((char*)d_ws + 8192);// [96] f32
    (void)ws_size;

    // 786432 floats = 196608 float4
    GRAPE_zero_k<<<768, 256, 0, stream>>>((float4*)out);
    GRAPE_fusew_k<<<96, 32, 0, stream>>>(w_enc, b_enc, w_head, b_head, Wf, bias);
    GRAPE_conv_raster_k<<<512, 256, 0, stream>>>(inp, Wf, bias, out);
    GRAPE_clamp_k<<<768, 256, 0, stream>>>((float4*)out);
}
